// Decoder_55877524521587
// MI455X (gfx1250) — matrix operand order repaired and checked by reading, not compiled, not run on hardware
//
#include <hip/hip_runtime.h>
#include <math.h>

#define HDIM 1024
#define BROWS 256
#define TSTEPS 128

typedef _Float16 v16h __attribute__((ext_vector_type(16)));
typedef _Float16 v8h  __attribute__((ext_vector_type(8)));
typedef float    v8f  __attribute__((ext_vector_type(8)));

__device__ __forceinline__ v8h ld8h(const _Float16* p) {
    return *reinterpret_cast<const v8h*>(p);
}

__device__ __forceinline__ v16h cat16(v8h lo, v8h hi) {
    return __builtin_shufflevector(lo, hi, 0,1,2,3,4,5,6,7,8,9,10,11,12,13,14,15);
}

__device__ __forceinline__ v8f wmma_f16(v16h a, v16h b, v8f c) {
    return __builtin_amdgcn_wmma_f32_16x16x32_f16(false, a, false, b, (short)0, c,
                                                  false, false);
}

// Computes a 32x32 tile of  A(f16,[BROWS x H]) @ W(f16,[H x H], row-major W[n][k])^T
// i.e. out[m,n] = sum_k A[m,k] * W[n,k], accumulated into acc[2][2] (f32).
// A-frag layout (16-bit A 16x32): lane = M row, halves 0-7: K=8*lg.., halves 8-15: K=16+8*lg..
// B-frag layout (16-bit B 32x16): lane = N col, halves 0-7: K=8*lg.., halves 8-15: K=16+8*lg..
__device__ __forceinline__ void gemm_tile32(const _Float16* __restrict__ A,
                                            const _Float16* __restrict__ W,
                                            int m0, int n0, int lane,
                                            v8f acc[2][2]) {
    const int lm = lane & 15;
    const int lg = lane >> 4;
    const _Float16* a0p = A + (size_t)(m0 + lm) * HDIM + 8 * lg;
    const _Float16* a1p = a0p + 16 * (size_t)HDIM;
    const _Float16* b0p = W + (size_t)(n0 + lm) * HDIM + 8 * lg;
    const _Float16* b1p = b0p + 16 * (size_t)HDIM;
#pragma unroll 4
    for (int k0 = 0; k0 < HDIM; k0 += 32) {
        v16h a0 = cat16(ld8h(a0p + k0), ld8h(a0p + k0 + 16));
        v16h a1 = cat16(ld8h(a1p + k0), ld8h(a1p + k0 + 16));
        v16h b0 = cat16(ld8h(b0p + k0), ld8h(b0p + k0 + 16));
        v16h b1 = cat16(ld8h(b1p + k0), ld8h(b1p + k0 + 16));
        acc[0][0] = wmma_f16(a0, b0, acc[0][0]);
        acc[0][1] = wmma_f16(a0, b1, acc[0][1]);
        acc[1][0] = wmma_f16(a1, b0, acc[1][0]);
        acc[1][1] = wmma_f16(a1, b1, acc[1][1]);
    }
}

// f32 -> f16 convert; sources are read exactly once -> non-temporal loads keep
// the 32 MB of f32 weight originals out of the L2 reuse set.
__global__ void __launch_bounds__(256)
cvt_kernel(_Float16* __restrict__ dst, const float* __restrict__ src, int n) {
    int i = blockIdx.x * blockDim.x + threadIdx.x;
    int stride = gridDim.x * blockDim.x;
    for (; i < n; i += stride) dst[i] = (_Float16)__builtin_nontemporal_load(src + i);
}

__global__ void __launch_bounds__(256)
zero_kernel(unsigned int* __restrict__ p, int nwords) {
    int i = blockIdx.x * blockDim.x + threadIdx.x;
    int stride = gridDim.x * blockDim.x;
    for (; i < nwords; i += stride) p[i] = 0u;
}

// Computes X[g] = x @ W_x[g]^T + b_x[g] for the 4 gates (g = wave id).
__global__ void __launch_bounds__(128)
xproj_kernel(const _Float16* __restrict__ Wx, const _Float16* __restrict__ xf16,
             float* __restrict__ X,
             const float* __restrict__ bxi, const float* __restrict__ bxf,
             const float* __restrict__ bxo, const float* __restrict__ bxc) {
    const int n0 = blockIdx.x * 32;
    const int m0 = blockIdx.y * 32;
    const int wave = threadIdx.x >> 5;
    const int lane = threadIdx.x & 31;
    v8f acc[2][2] = {};
    gemm_tile32(xf16, Wx + (size_t)wave * HDIM * HDIM, m0, n0, lane, acc);

    const float* bx = (wave == 0) ? bxi : (wave == 1) ? bxf : (wave == 2) ? bxo : bxc;
    float* Xg = X + (size_t)wave * BROWS * HDIM;
    const int lm = lane & 15;
    const int lg = lane >> 4;
#pragma unroll
    for (int i = 0; i < 2; i++)
#pragma unroll
        for (int j = 0; j < 2; j++) {
            int n = n0 + j * 16 + lm;
            float bv = bx[n];
#pragma unroll
            for (int r = 0; r < 8; r++) {
                int m = m0 + i * 16 + r + 8 * lg;
                Xg[(size_t)m * HDIM + n] = acc[i][j][r] + bv;
            }
        }
}

// One LSTM timestep: 4 gate GEMMs (h@W_h^T) + gates + c/h update.
__global__ void __launch_bounds__(128)
step_kernel(const _Float16* __restrict__ Wh, const _Float16* __restrict__ hin,
            _Float16* __restrict__ hout, float* __restrict__ C,
            const float* __restrict__ X,
            const float* __restrict__ bhi, const float* __restrict__ bhf,
            const float* __restrict__ bho, const float* __restrict__ bhc,
            float* __restrict__ out, int t) {
    __shared__ float lds[4][32][33];
    const int n0 = blockIdx.x * 32;
    const int m0 = blockIdx.y * 32;
    const int wave = threadIdx.x >> 5;
    const int lane = threadIdx.x & 31;

    v8f acc[2][2] = {};
    gemm_tile32(hin, Wh + (size_t)wave * HDIM * HDIM, m0, n0, lane, acc);

    const int lm = lane & 15;
    const int lg = lane >> 4;
#pragma unroll
    for (int i = 0; i < 2; i++)
#pragma unroll
        for (int j = 0; j < 2; j++)
#pragma unroll
            for (int r = 0; r < 8; r++)
                lds[wave][i * 16 + r + 8 * lg][j * 16 + lm] = acc[i][j][r];
    __syncthreads();

    const size_t BH = (size_t)BROWS * HDIM;
#pragma unroll
    for (int q = 0; q < 8; q++) {
        int e = q * 128 + threadIdx.x;   // 0..1023 within the 32x32 tile
        int ml = e >> 5;
        int nl = e & 31;
        int m = m0 + ml;
        int n = n0 + nl;
        size_t gi = (size_t)m * HDIM + n;

        float ip = lds[0][ml][nl] + X[gi]          + bhi[n];
        float fp = lds[1][ml][nl] + X[BH + gi]     + bhf[n];
        float op = lds[2][ml][nl] + X[2 * BH + gi] + bho[n];
        float gp = lds[3][ml][nl] + X[3 * BH + gi] + bhc[n];

        float it = 1.0f / (1.0f + __expf(-ip));
        float ft = 1.0f / (1.0f + __expf(-fp));
        float ot = 1.0f / (1.0f + __expf(-op));
        float gt = tanhf(gp);

        float cn = ft * C[gi] + it * gt;
        float hn = ot * tanhf(cn);

        C[gi] = cn;
        // Output slab is write-once / never re-read this call: non-temporal
        // store keeps the 134 MB stream from evicting the L2 weight set.
        __builtin_nontemporal_store(hn, out + ((size_t)m * TSTEPS + t) * HDIM + n);
        hout[gi] = (_Float16)hn;
    }
}

extern "C" void kernel_launch(void* const* d_in, const int* in_sizes, int n_in,
                              void* d_out, int out_size, void* d_ws, size_t ws_size,
                              hipStream_t stream) {
    const float* x    = (const float*)d_in[0];
    const float* W_xi = (const float*)d_in[1];  const float* b_xi = (const float*)d_in[2];
    const float* W_hi = (const float*)d_in[3];  const float* b_hi = (const float*)d_in[4];
    const float* W_xf = (const float*)d_in[5];  const float* b_xf = (const float*)d_in[6];
    const float* W_hf = (const float*)d_in[7];  const float* b_hf = (const float*)d_in[8];
    const float* W_xo = (const float*)d_in[9];  const float* b_xo = (const float*)d_in[10];
    const float* W_ho = (const float*)d_in[11]; const float* b_ho = (const float*)d_in[12];
    const float* W_xc = (const float*)d_in[13]; const float* b_xc = (const float*)d_in[14];
    const float* W_hc = (const float*)d_in[15]; const float* b_hc = (const float*)d_in[16];

    const size_t HH = (size_t)HDIM * HDIM;      // 1M
    const size_t BH = (size_t)BROWS * HDIM;     // 256K

    // Workspace layout (total ~22.5 MB):
    _Float16* Wh   = (_Float16*)d_ws;           // 4*HH f16 (gates i,f,o,c)  8 MB
    _Float16* Wx   = Wh + 4 * HH;               // 4*HH f16                  8 MB
    _Float16* xf16 = Wx + 4 * HH;               // BH f16                    0.5 MB
    _Float16* hbuf = xf16 + BH;                 // 2*BH f16 (double buffer)  1 MB
    float*    C    = (float*)(hbuf + 2 * BH);   // BH f32                    1 MB
    float*    X    = C + BH;                    // 4*BH f32                  4 MB

    // f32 -> f16 weight/x conversion (L2-resident from here on).
    cvt_kernel<<<2048, 256, 0, stream>>>(Wh + 0 * HH, W_hi, (int)HH);
    cvt_kernel<<<2048, 256, 0, stream>>>(Wh + 1 * HH, W_hf, (int)HH);
    cvt_kernel<<<2048, 256, 0, stream>>>(Wh + 2 * HH, W_ho, (int)HH);
    cvt_kernel<<<2048, 256, 0, stream>>>(Wh + 3 * HH, W_hc, (int)HH);
    cvt_kernel<<<2048, 256, 0, stream>>>(Wx + 0 * HH, W_xi, (int)HH);
    cvt_kernel<<<2048, 256, 0, stream>>>(Wx + 1 * HH, W_xf, (int)HH);
    cvt_kernel<<<2048, 256, 0, stream>>>(Wx + 2 * HH, W_xo, (int)HH);
    cvt_kernel<<<2048, 256, 0, stream>>>(Wx + 3 * HH, W_xc, (int)HH);
    cvt_kernel<<<512, 256, 0, stream>>>(xf16, x, (int)BH);
    // Zero h double-buffer (1 MB) + c state (1 MB): contiguous 524288 dwords.
    zero_kernel<<<1024, 256, 0, stream>>>((unsigned int*)hbuf, 524288);

    dim3 grid(HDIM / 32, BROWS / 32);   // 32 x 8 workgroups
    dim3 block(128);                    // 4 waves, one gate per wave

    // Loop-invariant x projections (bias folded in).
    xproj_kernel<<<grid, block, 0, stream>>>(Wx, xf16, X, b_xi, b_xf, b_xo, b_xc);

    float* out = (float*)d_out;
    for (int t = 0; t < TSTEPS; t++) {
        const _Float16* hin = hbuf + (size_t)(t & 1) * BH;
        _Float16* hout      = hbuf + (size_t)((t + 1) & 1) * BH;
        step_kernel<<<grid, block, 0, stream>>>(Wh, hin, hout, C, X,
                                                b_hi, b_hf, b_ho, b_hc, out, t);
    }
}
